// GraphEdgeEncoder_55173149884803
// MI455X (gfx1250) — compile-verified, matrix operand order checked by reading, not run on hardware
//
#include <hip/hip_runtime.h>

// ---------------- types for WMMA ----------------
typedef __attribute__((ext_vector_type(16))) __bf16 v16bf;
typedef __attribute__((ext_vector_type(8)))  float  v8f;

union FragBF { v16bf v; uint4 q[2]; };

__device__ __forceinline__ unsigned short f2bf(float x) {
    unsigned int u = __float_as_uint(x);
    u += 0x7fffu + ((u >> 16) & 1u);          // round-to-nearest-even
    return (unsigned short)(u >> 16);
}

#define Bsz 128
#define Nn  128
#define Ee  127
#define Dd  256
#define Hh  4
#define Mrows (Bsz * Nn)   // 16384

// ---------------- adjacency ----------------
__global__ void adj_init_kernel(float* adj) {
    int idx = blockIdx.x * blockDim.x + threadIdx.x;       // B*N*N threads
    int j = idx & (Nn - 1);
    int i = (idx >> 7) & (Nn - 1);
    adj[idx] = (i == j) ? 1.0f : 0.0f;
}

__global__ void adj_scatter_kernel(const int* __restrict__ edges, float* adj) {
    int idx = blockIdx.x * blockDim.x + threadIdx.x;       // B*E threads
    if (idx >= Bsz * Ee) return;
    int b = idx / Ee, e = idx - b * Ee;
    int src = edges[(b * Ee + e) * 2 + 0];
    int dst = edges[(b * Ee + e) * 2 + 1];
    adj[((size_t)b * Nn + src) * Nn + dst] = 1.0f;
    adj[((size_t)b * Nn + dst) * Nn + src] = 1.0f;
}

// ---------------- f32 -> bf16 converts ----------------
// node_features (M x 256) -> first half of combined (ld 512)
__global__ void conv_x_kernel(const float* __restrict__ x, unsigned short* __restrict__ comb) {
    int idx = blockIdx.x * blockDim.x + threadIdx.x;       // M*256 threads
    int row = idx >> 8, col = idx & 255;
    comb[(size_t)row * 512 + col] = f2bf(x[idx]);
}

// W (K x Nc) f32 -> Wt (Nc x K) bf16
__global__ void conv_wt_kernel(const float* __restrict__ w, unsigned short* __restrict__ wt,
                               int K, int Nc) {
    int idx = blockIdx.x * blockDim.x + threadIdx.x;
    if (idx >= K * Nc) return;
    int n = idx / K, k = idx - n * K;
    wt[(size_t)n * K + k] = f2bf(w[(size_t)k * Nc + n]);
}

// ---------------- generic bf16 WMMA GEMM, 16x64 strip per wave ----------------
// C[M x N] = act( A[M x K](bf16, lda) @ Bt[N x K](bf16)^T + bias ) (+ residual)
// Each wave: one 16-row x 64-col strip = 4 WMMA accumulators sharing the A fragment.
__global__ __launch_bounds__(256)
void gemm_bf16_kernel(const unsigned short* __restrict__ A, int lda,
                      const unsigned short* __restrict__ Bt,
                      const float* __restrict__ bias,
                      float* Cf, unsigned short* Cb, int ldc,
                      const float* __restrict__ residual,
                      int M, int N, int K, int do_relu) {
    int wid  = threadIdx.x >> 5;
    int lane = threadIdx.x & 31;
    int tile = blockIdx.x * 8 + wid;
    int ntn  = N >> 6;                              // strips of 64 columns
    if (tile >= (M >> 4) * ntn) return;
    int mt = tile / ntn, nt = tile - mt * ntn;
    int half = lane >> 4, r = lane & 15;

    const unsigned short* arow = A + (size_t)(mt * 16 + r) * lda;
    const unsigned short* brow[4];
#pragma unroll
    for (int j = 0; j < 4; ++j)
        brow[j] = Bt + (size_t)(nt * 64 + j * 16 + r) * K;

    v8f c[4];
#pragma unroll
    for (int j = 0; j < 4; ++j) c[j] = (v8f){0.f,0.f,0.f,0.f,0.f,0.f,0.f,0.f};

    for (int kk = 0; kk < K; kk += 32) {
        int khi = kk + half * 8;
        FragBF a;
        a.q[0] = *(const uint4*)(arow + khi);
        a.q[1] = *(const uint4*)(arow + khi + 16);
#pragma unroll
        for (int j = 0; j < 4; ++j) {
            FragBF b;
            b.q[0] = *(const uint4*)(brow[j] + khi);
            b.q[1] = *(const uint4*)(brow[j] + khi + 16);
            c[j] = __builtin_amdgcn_wmma_f32_16x16x32_bf16(false, a.v, false, b.v,
                                                           (short)0, c[j], false, false);
        }
    }
#pragma unroll
    for (int j = 0; j < 4; ++j) {
        int n = nt * 64 + j * 16 + r;
        float bv = bias ? bias[n] : 0.f;
#pragma unroll
        for (int jv = 0; jv < 8; ++jv) {
            int m = mt * 16 + half * 8 + jv;
            float val = c[j][jv] + bv;
            if (do_relu) val = fmaxf(val, 0.f);
            if (residual) val += residual[(size_t)m * 256 + n];
            if (Cf) Cf[(size_t)m * ldc + n] = val;
            if (Cb) Cb[(size_t)m * ldc + n] = f2bf(val);
        }
    }
}

// ---------------- transpose V for the PV matmul ----------------
// qkv (M x 768 bf16), v part cols [512,768) -> vT[(b,h,d,j)]  (B*H*64*128)
__global__ void transpose_v_kernel(const unsigned short* __restrict__ qkv,
                                   unsigned short* __restrict__ vT) {
    int idx = blockIdx.x * blockDim.x + threadIdx.x;       // B*H*64*128 threads
    int j = idx & 127;
    int d = (idx >> 7) & 63;
    int h = (idx >> 13) & 3;
    int b = idx >> 15;
    vT[idx] = qkv[((size_t)(b * Nn + j)) * 768 + 512 + h * 64 + d];
}

// ---------------- fused masked attention ----------------
// one wave = one 16-row q-tile of one (b,h); scores->softmax->ctx, all WMMA
__global__ __launch_bounds__(256)
void attn_kernel(const unsigned short* __restrict__ qkv,
                 const unsigned short* __restrict__ vT,
                 const float* __restrict__ adj,
                 unsigned short* __restrict__ ctx) {
    __shared__ unsigned short probs[8][16 * 128];          // 32 KB, 4KB/wave
    int wid  = threadIdx.x >> 5;
    int lane = threadIdx.x & 31;
    int gw = blockIdx.x * 8 + wid;                         // B*H*8 waves total
    int it = gw & 7, h = (gw >> 3) & 3, b = gw >> 5;
    int half = lane >> 4, r = lane & 15;

    // ---- scores = q @ k^T  (16 x 128), dh = 64 ----
    v8f s[8];
#pragma unroll
    for (int jt = 0; jt < 8; ++jt) s[jt] = (v8f){0.f,0.f,0.f,0.f,0.f,0.f,0.f,0.f};

    const unsigned short* qrow = qkv + (size_t)(b * Nn + it * 16 + r) * 768 + h * 64;
    for (int kk = 0; kk < 64; kk += 32) {
        int khi = kk + half * 8;
        FragBF a;
        a.q[0] = *(const uint4*)(qrow + khi);
        a.q[1] = *(const uint4*)(qrow + khi + 16);
#pragma unroll
        for (int jt = 0; jt < 8; ++jt) {
            const unsigned short* krow =
                qkv + (size_t)(b * Nn + jt * 16 + r) * 768 + 256 + h * 64;
            FragBF bb;
            bb.q[0] = *(const uint4*)(krow + khi);
            bb.q[1] = *(const uint4*)(krow + khi + 16);
            s[jt] = __builtin_amdgcn_wmma_f32_16x16x32_bf16(false, a.v, false, bb.v,
                                                            (short)0, s[jt], false, false);
        }
    }

    // ---- mask + scale, rowwise softmax (rows split: half0 -> rows 0-7, half1 -> 8-15) ----
    float mx[8], sum[8];
#pragma unroll
    for (int jv = 0; jv < 8; ++jv) {
        int row = it * 16 + half * 8 + jv;
        float m = -INFINITY;
#pragma unroll
        for (int jt = 0; jt < 8; ++jt) {
            int col = jt * 16 + r;
            float sv = s[jt][jv] * 0.125f;                 // 1/sqrt(64)
            bool ok = adj[((size_t)b * Nn + row) * Nn + col] > 0.f;
            sv = ok ? sv : -INFINITY;
            s[jt][jv] = sv;
            m = fmaxf(m, sv);
        }
#pragma unroll
        for (int off = 8; off >= 1; off >>= 1) m = fmaxf(m, __shfl_xor(m, off, 32));
        mx[jv] = m;
    }
#pragma unroll
    for (int jv = 0; jv < 8; ++jv) {
        float acc = 0.f;
#pragma unroll
        for (int jt = 0; jt < 8; ++jt) {
            float e = __expf(s[jt][jv] - mx[jv]);
            s[jt][jv] = e;
            acc += e;
        }
#pragma unroll
        for (int off = 8; off >= 1; off >>= 1) acc += __shfl_xor(acc, off, 32);
        sum[jv] = acc;
    }
#pragma unroll
    for (int jv = 0; jv < 8; ++jv) {
        float inv = 1.0f / sum[jv];
#pragma unroll
        for (int jt = 0; jt < 8; ++jt)
            probs[wid][(half * 8 + jv) * 128 + jt * 16 + r] = f2bf(s[jt][jv] * inv);
    }
    __syncthreads();

    // ---- ctx = probs (16x128) @ v (128x64) ----
    v8f cf[4];
#pragma unroll
    for (int dt = 0; dt < 4; ++dt) cf[dt] = (v8f){0.f,0.f,0.f,0.f,0.f,0.f,0.f,0.f};
    for (int kk = 0; kk < 128; kk += 32) {
        int khi = kk + half * 8;
        FragBF a;
        a.q[0] = *(const uint4*)&probs[wid][r * 128 + khi];
        a.q[1] = *(const uint4*)&probs[wid][r * 128 + khi + 16];
#pragma unroll
        for (int dt = 0; dt < 4; ++dt) {
            const unsigned short* vrow =
                vT + ((size_t)(b * Hh + h) * 64 + dt * 16 + r) * 128;
            FragBF bb;
            bb.q[0] = *(const uint4*)(vrow + khi);
            bb.q[1] = *(const uint4*)(vrow + khi + 16);
            cf[dt] = __builtin_amdgcn_wmma_f32_16x16x32_bf16(false, a.v, false, bb.v,
                                                             (short)0, cf[dt], false, false);
        }
    }
    // store ctx bf16 at (b*128+i, h*64+d) in M x 256
#pragma unroll
    for (int dt = 0; dt < 4; ++dt)
#pragma unroll
        for (int jv = 0; jv < 8; ++jv) {
            int m = b * Nn + it * 16 + half * 8 + jv;
            int col = h * 64 + dt * 16 + r;
            ctx[(size_t)m * 256 + col] = f2bf(cf[dt][jv]);
        }
}

// ---------------- layernorm (wave per row) ----------------
__global__ __launch_bounds__(256)
void ln_kernel(const float* __restrict__ x, const float* __restrict__ g,
               const float* __restrict__ be, float* __restrict__ out) {
    int wid = threadIdx.x >> 5, lane = threadIdx.x & 31;
    int row = blockIdx.x * 8 + wid;
    const float* xr = x + (size_t)row * 256;
    float4 v0 = ((const float4*)xr)[lane * 2];
    float4 v1 = ((const float4*)xr)[lane * 2 + 1];
    float s  = v0.x + v0.y + v0.z + v0.w + v1.x + v1.y + v1.z + v1.w;
    float ss = v0.x*v0.x + v0.y*v0.y + v0.z*v0.z + v0.w*v0.w
             + v1.x*v1.x + v1.y*v1.y + v1.z*v1.z + v1.w*v1.w;
#pragma unroll
    for (int off = 16; off >= 1; off >>= 1) {
        s  += __shfl_xor(s,  off, 32);
        ss += __shfl_xor(ss, off, 32);
    }
    float mean = s * (1.0f / 256.0f);
    float var  = ss * (1.0f / 256.0f) - mean * mean;
    float inv  = rsqrtf(var + 1e-5f);
    float xs[8] = {v0.x, v0.y, v0.z, v0.w, v1.x, v1.y, v1.z, v1.w};
#pragma unroll
    for (int i = 0; i < 8; ++i) {
        int col = lane * 8 + i;
        out[(size_t)row * 256 + col] = g[col] * (xs[i] - mean) * inv + be[col];
    }
}

// ---------------- host launch ----------------
extern "C" void kernel_launch(void* const* d_in, const int* in_sizes, int n_in,
                              void* d_out, int out_size, void* d_ws, size_t ws_size,
                              hipStream_t stream) {
    const float* node  = (const float*)d_in[0];
    const int*   edges = (const int*)d_in[2];
    const float* in_w  = (const float*)d_in[7];
    const float* in_b  = (const float*)d_in[8];
    const float* out_w = (const float*)d_in[9];
    const float* out_b = (const float*)d_in[10];
    const float* nu_w1 = (const float*)d_in[11];
    const float* nu_b1 = (const float*)d_in[12];
    const float* nu_w2 = (const float*)d_in[13];
    const float* nu_b2 = (const float*)d_in[14];
    const float* ln_g  = (const float*)d_in[15];
    const float* ln_b  = (const float*)d_in[16];

    char* ws = (char*)d_ws;
    const size_t OFF_ADJ  = 0;                              // 8 MB f32
    const size_t OFF_COMB = OFF_ADJ  + (size_t)Bsz*Nn*Nn*4; // 16 MB bf16 (M x 512)
    const size_t OFF_QKV  = OFF_COMB + (size_t)Mrows*512*2; // 24 MB bf16 (M x 768)
    const size_t OFF_VT   = OFF_QKV  + (size_t)Mrows*768*2; // 8 MB bf16
    const size_t OFF_CTX  = OFF_VT   + (size_t)Bsz*Hh*64*128*2; // 8 MB bf16
    const size_t OFF_WT   = OFF_CTX  + (size_t)Mrows*256*2;
    const size_t OFF_INWT = OFF_WT;                         // 768x256
    const size_t OFF_OUWT = OFF_INWT + (size_t)768*256*2;   // 256x256
    const size_t OFF_N1WT = OFF_OUWT + (size_t)256*256*2;   // 256x512
    const size_t OFF_N2WT = OFF_N1WT + (size_t)256*512*2;   // 256x256
    // reuse qkv space after attention:
    const size_t OFF_U1 = OFF_QKV;                          // 8 MB bf16 (M x 256)
    const size_t OFF_X  = OFF_QKV + (size_t)Mrows*256*2;    // 16 MB f32 (M x 256)

    float*          adj  = (float*)(ws + OFF_ADJ);
    unsigned short* comb = (unsigned short*)(ws + OFF_COMB);
    unsigned short* qkvb = (unsigned short*)(ws + OFF_QKV);
    unsigned short* vT   = (unsigned short*)(ws + OFF_VT);
    unsigned short* ctxb = (unsigned short*)(ws + OFF_CTX);
    unsigned short* inwT = (unsigned short*)(ws + OFF_INWT);
    unsigned short* ouwT = (unsigned short*)(ws + OFF_OUWT);
    unsigned short* n1wT = (unsigned short*)(ws + OFF_N1WT);
    unsigned short* n2wT = (unsigned short*)(ws + OFF_N2WT);
    unsigned short* u1b  = (unsigned short*)(ws + OFF_U1);
    float*          xbuf = (float*)(ws + OFF_X);

    // 1. adjacency
    adj_init_kernel<<<(Bsz*Nn*Nn)/256, 256, 0, stream>>>(adj);
    adj_scatter_kernel<<<(Bsz*Ee + 255)/256, 256, 0, stream>>>(edges, adj);
    // 2. converts
    conv_x_kernel<<<(Mrows*256)/256, 256, 0, stream>>>(node, comb);
    conv_wt_kernel<<<(256*768 + 255)/256, 256, 0, stream>>>(in_w,  inwT, 256, 768);
    conv_wt_kernel<<<(256*256 + 255)/256, 256, 0, stream>>>(out_w, ouwT, 256, 256);
    conv_wt_kernel<<<(512*256 + 255)/256, 256, 0, stream>>>(nu_w1, n1wT, 512, 256);
    conv_wt_kernel<<<(256*256 + 255)/256, 256, 0, stream>>>(nu_w2, n2wT, 256, 256);
    // 3. qkv = x @ in_w + in_b   (M=16384 N=768 K=256), bf16 out
    gemm_bf16_kernel<<<(Mrows/16)*(768/64)/8, 256, 0, stream>>>(
        comb, 512, inwT, in_b, nullptr, qkvb, 768, nullptr, Mrows, 768, 256, 0);
    // 4. v transpose
    transpose_v_kernel<<<(Bsz*Hh*64*128)/256, 256, 0, stream>>>(qkvb, vT);
    // 5. attention -> ctx bf16
    attn_kernel<<<(Bsz*Hh*8)/8, 256, 0, stream>>>(qkvb, vT, adj, ctxb);
    // 6. graph_features = ctx @ out_w + out_b -> second half of comb (bf16, ld 512)
    gemm_bf16_kernel<<<(Mrows/16)*(256/64)/8, 256, 0, stream>>>(
        ctxb, 256, ouwT, out_b, nullptr, comb + 256, 512, nullptr, Mrows, 256, 256, 0);
    // 7. u1 = relu(comb @ nu_w1 + nu_b1)   (K=512), bf16 out
    gemm_bf16_kernel<<<(Mrows/16)*(256/64)/8, 256, 0, stream>>>(
        comb, 512, n1wT, nu_b1, nullptr, u1b, 256, nullptr, Mrows, 256, 512, 1);
    // 8. x = u1 @ nu_w2 + nu_b2 + node   (f32 out)
    gemm_bf16_kernel<<<(Mrows/16)*(256/64)/8, 256, 0, stream>>>(
        u1b, 256, n2wT, nu_b2, xbuf, nullptr, 256, node, Mrows, 256, 256, 0);
    // 9. layernorm -> d_out
    ln_kernel<<<Mrows/8, 256, 0, stream>>>(xbuf, ln_g, ln_b, (float*)d_out);
}
